// PatchEmbedding_40106404610305
// MI455X (gfx1250) — matrix operand order repaired and checked by reading, not compiled, not run on hardware
//
#include <hip/hip_runtime.h>
#include <math.h>

#define IN_C 3
#define DD   96
#define PP   4
#define IMG  640
#define BB   2
#define HP   160
#define NTOK (HP*HP)      // 25600 tokens per batch
#define HW   (IMG*IMG)    // 409600

typedef __attribute__((ext_vector_type(16))) __bf16        v16bf;
typedef __attribute__((ext_vector_type(8)))  float         v8f;
typedef __attribute__((ext_vector_type(4)))  unsigned int  v4u;

union Frag { v16bf bf; v4u u[2]; };

__device__ __forceinline__ unsigned short f2bf(float f) {
    unsigned int u = __float_as_uint(f);
    unsigned int r = u + 0x7FFFu + ((u >> 16) & 1u);   // round-to-nearest-even
    return (unsigned short)(r >> 16);
}
__device__ __forceinline__ float bf2f(unsigned short h) {
    return __uint_as_float(((unsigned int)h) << 16);
}

// ---------------------------------------------------------------------------
// Kernel 1: 1x1 conv (3 -> 96), NCHW input -> NHWC fp32 + NHWC bf16
// block = 96 threads (3 waves), one thread per output channel, 8 pixels/block
// ---------------------------------------------------------------------------
__global__ void __launch_bounds__(96)
k_embed(const float* __restrict__ x, const float* __restrict__ w,
        const float* __restrict__ bias,
        float* __restrict__ xe_f32, unsigned short* __restrict__ xe_bf)
{
    int d = threadIdx.x;                       // 0..95
    long pix0 = (long)blockIdx.x * 8;
    float w0 = w[d*3+0], w1 = w[d*3+1], w2 = w[d*3+2];
    float bd = bias[d];
#pragma unroll
    for (int p = 0; p < 8; ++p) {
        long pix = pix0 + p;
        if (pix >= (long)BB*HW) break;
        int b  = (int)(pix / HW);
        int hw = (int)(pix - (long)b*HW);
        float v = bd + x[((long)(b*3+0))*HW + hw] * w0
                     + x[((long)(b*3+1))*HW + hw] * w1
                     + x[((long)(b*3+2))*HW + hw] * w2;
        long o = pix*DD + d;
        xe_f32[o] = v;
        xe_bf[o]  = f2bf(v);
    }
}

// ---------------------------------------------------------------------------
// Kernel 2: pack ph_w1 (OIHW 96x96x3x3) into WMMA B-matrix (32x16 bf16) lane
// layout: frag[tap][cChunk][nChunk], within frag: lane*16 + half,
//   N = nChunk*16 + (lane&15),  K = (lane>=16 ? 8 : 0) + (half&7) + (half>=8 ? 16 : 0), c = cChunk*32+K
// ---------------------------------------------------------------------------
__global__ void k_pack_w1(const float* __restrict__ w1,
                          unsigned short* __restrict__ wp)
{
    int i = blockIdx.x * blockDim.x + threadIdx.x;
    const int TOT = 9*3*6*32*16;               // 82944
    if (i >= TOT) return;
    int j    =  i        & 15;
    int lane = (i >>  4) & 31;
    int nc   = (i >>  9) % 6;
    int cc   = (i / 3072) % 3;
    int tap  =  i / 9216;
    int n  = nc*16 + (lane & 15);
    int c  = cc*32 + ((lane & 16) ? 8 : 0) + (j & 7) + ((j & 8) ? 16 : 0);
    int ky = tap / 3, kx = tap % 3;
    wp[i] = f2bf(w1[((n*DD + c)*3 + ky)*3 + kx]);
}

// ---------------------------------------------------------------------------
// Kernel 3: 3x3 SAME conv 96->96 with bf16 WMMA + bias + exact GELU -> bf16
// One wave per 32 pixels (two 16-pixel M-tiles) x all 96 output channels.
// Each B fragment (weights) is reused by TWO WMMAs -> halves L2 weight
// traffic vs one-tile blocking.  9 taps x 3 K-chunks x 6 N-chunks x 2 tiles
// = 324 v_wmma_f32_16x16x32_bf16 per wave.
// ---------------------------------------------------------------------------
__global__ void __launch_bounds__(256)
k_conv1(const unsigned short* __restrict__ xe_bf,
        const unsigned short* __restrict__ wp,
        const float* __restrict__ b1,
        unsigned short* __restrict__ h_bf)
{
    int wave = (blockIdx.x << 3) + (threadIdx.x >> 5);   // 8 waves / block
    int lane = threadIdx.x & 31;
    int b   = wave / (IMG*20);
    int rem = wave % (IMG*20);
    int y   = rem / 20;
    int x0  = (rem % 20) << 5;             // 32 pixels per wave

    int m    = lane & 15;                  // A-row = pixel within tile
    int off0 = (lane & 16) ? 8 : 0;        // K-offset for this lane group

    v8f acc0[6], acc1[6];
#pragma unroll
    for (int nc = 0; nc < 6; ++nc) {
        v8f z;
#pragma unroll
        for (int r = 0; r < 8; ++r) z[r] = 0.0f;
        acc0[nc] = z; acc1[nc] = z;
    }

#pragma unroll
    for (int tap = 0; tap < 9; ++tap) {
        int ky = tap/3 - 1, kx = tap%3 - 1;
        int yy  = y + ky;
        int xxA = x0 + m + kx;             // tile 0 pixel
        int xxB = xxA + 16;                // tile 1 pixel
        bool rowOk = (yy >= 0) & (yy < IMG);
        bool va = rowOk & (xxA >= 0) & (xxA < IMG);
        bool vb = rowOk & (xxB < IMG);     // xxB >= 15 always
        long pixA = ((long)(b*IMG + yy)*IMG + xxA) * DD;
        long pixB = pixA + 16*DD;

        // counter-free prefetch of the next row's A data (global_prefetch)
        if (va & (yy + 1 < IMG))
            __builtin_prefetch((const void*)(xe_bf + pixA + (long)IMG*DD), 0, 1);

#pragma unroll
        for (int cc = 0; cc < 3; ++cc) {
            // A fragments: 16-bit A 16x32 layout (halves 0-7 -> K=off0..off0+7,
            // halves 8-15 -> K=off0+16..off0+23)
            Frag a0, a1;
            v4u zv; zv.x = zv.y = zv.z = zv.w = 0u;
            if (va) {
                const unsigned short* p = xe_bf + pixA + cc*32 + off0;
                a0.u[0] = *(const v4u*)(p);
                a0.u[1] = *(const v4u*)(p + 16);
            } else { a0.u[0] = zv; a0.u[1] = zv; }
            if (vb) {
                const unsigned short* p = xe_bf + pixB + cc*32 + off0;
                a1.u[0] = *(const v4u*)(p);
                a1.u[1] = *(const v4u*)(p + 16);
            } else { a1.u[0] = zv; a1.u[1] = zv; }

#pragma unroll
            for (int nc = 0; nc < 6; ++nc) {
                const unsigned short* q =
                    wp + ((size_t)((tap*3 + cc)*6 + nc))*512 + lane*16;
                Frag bw;
                bw.u[0] = *(const v4u*)(q);
                bw.u[1] = *(const v4u*)(q + 8);
                acc0[nc] = __builtin_amdgcn_wmma_f32_16x16x32_bf16(
                    false, a0.bf, false, bw.bf, (short)0, acc0[nc], false, false);
                acc1[nc] = __builtin_amdgcn_wmma_f32_16x16x32_bf16(
                    false, a1.bf, false, bw.bf, (short)0, acc1[nc], false, false);
            }
        }
    }

    // Epilogue: D layout -> VGPR r holds pixel M = r + 8*(lane>=16), N = lane&15
    long rowBase = ((long)(b*IMG + y)*IMG) * DD;
    int  pxHi = (lane & 16) ? 8 : 0;
#pragma unroll
    for (int nc = 0; nc < 6; ++nc) {
        int ch = nc*16 + (lane & 15);
        float bias = b1[ch];
#pragma unroll
        for (int r = 0; r < 8; ++r) {
            float v0 = acc0[nc][r] + bias;
            v0 = 0.5f * v0 * (1.0f + erff(v0 * 0.70710678118654752f));
            h_bf[rowBase + (long)(x0 + pxHi + r)*DD + ch] = f2bf(v0);
            float v1 = acc1[nc][r] + bias;
            v1 = 0.5f * v1 * (1.0f + erff(v1 * 0.70710678118654752f));
            h_bf[rowBase + (long)(x0 + 16 + pxHi + r)*DD + ch] = f2bf(v1);
        }
    }
}

// ---------------------------------------------------------------------------
// Kernel 4: 4x4 stride-4 conv 96->3 (+bias). Weights staged in LDS.
// One thread per (b, py, px), computes all 3 output channels.
// ---------------------------------------------------------------------------
__global__ void __launch_bounds__(256)
k_conv2(const unsigned short* __restrict__ h_bf,
        const float* __restrict__ w2, const float* __restrict__ b2,
        float* __restrict__ prm)
{
    __shared__ float sw[3*96*16];
    for (int i = threadIdx.x; i < 3*96*16; i += blockDim.x) sw[i] = w2[i];
    __syncthreads();

    int t = blockIdx.x * blockDim.x + threadIdx.x;
    if (t >= BB*NTOK) return;
    int b  = t / NTOK;
    int r  = t % NTOK;
    int py = r / HP, px = r % HP;

    float s0 = 0.f, s1 = 0.f, s2 = 0.f;
#pragma unroll
    for (int iy = 0; iy < 4; ++iy)
#pragma unroll
    for (int ix = 0; ix < 4; ++ix) {
        long base = ((long)(b*IMG + py*4 + iy)*IMG + px*4 + ix) * DD;
        for (int c = 0; c < DD; ++c) {
            float hv = bf2f(h_bf[base + c]);
            int wi = (c*4 + iy)*4 + ix;
            s0 += hv * sw[wi];
            s1 += hv * sw[1536 + wi];
            s2 += hv * sw[3072 + wi];
        }
    }
    prm[(long)(b*3+0)*NTOK + r] = s0 + b2[0];
    prm[(long)(b*3+1)*NTOK + r] = s1 + b2[1];
    prm[(long)(b*3+2)*NTOK + r] = s2 + b2[2];
}

// ---------------------------------------------------------------------------
// Kernel 5: box params -> 16-pt bilinear sample of x_embed -> mean -> LN.
// One block per token, one thread per channel (96 active of 128).
// Also writes scale_map.
// ---------------------------------------------------------------------------
__global__ void __launch_bounds__(128)
k_token(const float* __restrict__ xe_f32, const float* __restrict__ prm,
        const float* __restrict__ ln_w, const float* __restrict__ ln_b,
        float* __restrict__ out)
{
    int t  = blockIdx.x;            // 0 .. BB*NTOK-1
    int b  = t / NTOK;
    int r  = t % NTOK;
    int py = r / HP, px = r % HP;

    float dxv = prm[(long)(b*3+0)*NTOK + r];
    float dyv = prm[(long)(b*3+1)*NTOK + r];
    float lsv = prm[(long)(b*3+2)*NTOK + r];
    float mx = 2.0f * tanhf(dxv);
    float my = 2.0f * tanhf(dyv);
    float s  = fminf(fmaxf(expf(tanhf(lsv)), 0.5f), 2.0f);
    float cx = px*4 + 2.0f, cy = py*4 + 2.0f;
    float half = 2.0f * s;
    float X1 = fminf(fmaxf(cx + mx - half, 0.f), (float)(IMG-1));
    float Y1 = fminf(fmaxf(cy + my - half, 0.f), (float)(IMG-1));
    float X2 = fminf(fmaxf(cx + mx + half, 0.f), (float)(IMG-1));
    float Y2 = fminf(fmaxf(cy + my + half, 0.f), (float)(IMG-1));

    int d = threadIdx.x;
    float accv = 0.f;
    if (d < DD) {
        const float* feat = xe_f32 + (long)b*HW*DD + d;
#pragma unroll
        for (int j = 0; j < 4; ++j) {
            float yv = Y1 + (Y2 - Y1) * (j * (1.0f/3.0f));
            float yf = floorf(yv);
            float wy = yv - yf;
            int y0i = min(max((int)yf, 0), IMG-1);
            int y1i = min(y0i + 1, IMG-1);
#pragma unroll
            for (int i = 0; i < 4; ++i) {
                float xv = X1 + (X2 - X1) * (i * (1.0f/3.0f));
                float xf = floorf(xv);
                float wx = xv - xf;
                int x0i = min(max((int)xf, 0), IMG-1);
                int x1i = min(x0i + 1, IMG-1);
                float v00 = feat[((long)y0i*IMG + x0i)*DD];
                float v01 = feat[((long)y0i*IMG + x1i)*DD];
                float v10 = feat[((long)y1i*IMG + x0i)*DD];
                float v11 = feat[((long)y1i*IMG + x1i)*DD];
                accv += v00*(1.f-wx)*(1.f-wy) + v01*wx*(1.f-wy)
                      + v10*(1.f-wx)*wy       + v11*wx*wy;
            }
        }
        accv *= (1.0f/16.0f);
    }

    __shared__ float sb[DD];
    if (d < DD) sb[d] = accv;
    __syncthreads();
    if (d < DD) {
        float mu = 0.f;
        for (int i = 0; i < DD; ++i) mu += sb[i];
        mu *= (1.0f/DD);
        float var = 0.f;
        for (int i = 0; i < DD; ++i) { float df = sb[i] - mu; var += df*df; }
        var *= (1.0f/DD);
        float rstd = rsqrtf(var + 1e-5f);
        out[(long)t*DD + d] = (accv - mu) * rstd * ln_w[d] + ln_b[d];
    }
    if (threadIdx.x == 0)
        out[(long)BB*NTOK*DD + t] = s;      // scale_map, flat (B,1,HP,HP)
}

// ---------------------------------------------------------------------------
extern "C" void kernel_launch(void* const* d_in, const int* in_sizes, int n_in,
                              void* d_out, int out_size, void* d_ws, size_t ws_size,
                              hipStream_t stream)
{
    const float* x       = (const float*)d_in[0];
    const float* embed_w = (const float*)d_in[1];
    const float* embed_b = (const float*)d_in[2];
    const float* ph_w1   = (const float*)d_in[3];
    const float* ph_b1   = (const float*)d_in[4];
    const float* ph_w2   = (const float*)d_in[5];
    const float* ph_b2   = (const float*)d_in[6];
    const float* ln_w    = (const float*)d_in[7];
    const float* ln_b    = (const float*)d_in[8];
    float* out = (float*)d_out;

    char* ws = (char*)d_ws;
    size_t o = 0;
    float*          xe_f32 = (float*)(ws + o);          o += (size_t)BB*HW*DD*4; // 314.6 MB
    unsigned short* xe_bf  = (unsigned short*)(ws + o); o += (size_t)BB*HW*DD*2; // 157.3 MB
    unsigned short* h_bf   = (unsigned short*)(ws + o); o += (size_t)BB*HW*DD*2; // 157.3 MB
    unsigned short* wp     = (unsigned short*)(ws + o); o += (size_t)9*3*6*32*16*2;
    o = (o + 255) & ~(size_t)255;
    float*          prm    = (float*)(ws + o);                                   // 614 KB

    k_embed  <<<(BB*HW + 7)/8, 96, 0, stream>>>(x, embed_w, embed_b, xe_f32, xe_bf);
    k_pack_w1<<<(9*3*6*32*16 + 255)/256, 256, 0, stream>>>(ph_w1, wp);
    k_conv1  <<<(BB*IMG*20)/8, 256, 0, stream>>>(xe_bf, wp, ph_b1, h_bf);
    k_conv2  <<<(BB*NTOK + 255)/256, 256, 0, stream>>>(h_bf, ph_w2, ph_b2, prm);
    k_token  <<<BB*NTOK, 128, 0, stream>>>(xe_f32, prm, ln_w, ln_b, out);
}
